// UltraModularDecoder_50964081935064
// MI455X (gfx1250) — matrix operand order repaired and checked by reading, not compiled, not run on hardware
//
#include <hip/hip_runtime.h>
#include <hip/hip_bf16.h>
#include <math.h>

typedef __attribute__((ext_vector_type(8)))  _Float16 v8h;
typedef __attribute__((ext_vector_type(16))) _Float16 v16h;
typedef __attribute__((ext_vector_type(8)))  float    v8f;
typedef __attribute__((ext_vector_type(4)))  unsigned int u32x4;
typedef __attribute__((ext_vector_type(8)))  int      i32x8;
typedef __attribute__((ext_vector_type(4)))  int      i32x4;

#define D_MODEL 1024
#define N_LAYERS 6
#define VOCAB   50257
#define SEQ     2048
#define D_FF    4096

#define TM 128
#define TN 128
#define TK 32

// ---------------------------------------------------------------------------
// Tensor Data Mover: DMA a 2D tile (tile_d0 x tile_d1 elements of 2 bytes,
// row stride = stride_elems) from global memory into packed LDS rows.
// D# bit layout per CDNA5 ISA (groups 0/1; groups 2/3
// zero-filled, 2D tile => tile_dim2 = 0). Issued once (wave 0); tracked by
// TENSORcnt. 6-arg builtin form (clang-23 / therock-10.0 headers).
// ---------------------------------------------------------------------------
__device__ __forceinline__ void tdm_load_tile_f16(
    unsigned lds_byte_addr, const void* gptr, unsigned tensor_d0,
    unsigned tensor_d1, unsigned tile_d0, unsigned tile_d1,
    unsigned stride_elems) {
  const unsigned long long ga = (unsigned long long)(uintptr_t)gptr;
  u32x4 g0;
  g0[0] = 1u;                                        // count=1 (valid), user D#
  g0[1] = lds_byte_addr;                             // lds_addr [63:32]
  g0[2] = (unsigned)(ga & 0xFFFFFFFFu);              // global_addr [95:64]
  g0[3] = (unsigned)((ga >> 32) & 0x01FFFFFFu)       // global_addr [120:96]
          | (2u << 30);                              // type=2 ("image") [127:126]
  i32x8 g1;
  g1[0] = (int)(1u << 16);                           // data_size=1 (2 bytes)
  g1[1] = (int)((tensor_d0 & 0xFFFFu) << 16);        // tensor_dim0 [79:48]
  g1[2] = (int)(((tensor_d0 >> 16) & 0xFFFFu) |
                ((tensor_d1 & 0xFFFFu) << 16));      // tensor_dim0/1
  g1[3] = (int)(((tensor_d1 >> 16) & 0xFFFFu) |
                ((tile_d0 & 0xFFFFu) << 16));        // tensor_dim1 / tile_dim0
  g1[4] = (int)(tile_d1 & 0xFFFFu);                  // tile_dim1 (tile_dim2=0)
  g1[5] = (int)stride_elems;                         // tensor_dim0_stride lo
  g1[6] = 0;                                         // stride hi / dim1_stride lo
  g1[7] = 0;
  const i32x4 gz4 = {0, 0, 0, 0};
  const i32x8 gz8 = {0, 0, 0, 0, 0, 0, 0, 0};
  __builtin_amdgcn_tensor_load_to_lds(g0, g1, gz4, gz4, gz8, 0);
}

// ---------------------------------------------------------------------------
// Embedding gather: x_f16[s][d] = (f16) emb[ids[s]][d]
// ---------------------------------------------------------------------------
__global__ __launch_bounds__(256) void embed_gather_f16(
    const int* __restrict__ ids, const float* __restrict__ emb,
    _Float16* __restrict__ x) {
  const int s = blockIdx.x;
  const int row = ids[s];
  const float* src = emb + (size_t)row * D_MODEL;
  _Float16* dst = x + (size_t)s * D_MODEL;
  for (int d = threadIdx.x; d < D_MODEL; d += 256)
    dst[d] = (_Float16)src[d];
}

// ---------------------------------------------------------------------------
// Row softmax: probs_f16 = softmax(scores_f32) along last dim (n columns)
// ---------------------------------------------------------------------------
__global__ __launch_bounds__(256) void softmax_rows_f16(
    const float* __restrict__ S, _Float16* __restrict__ P, int n) {
  __shared__ float red[256];
  const int row = blockIdx.x;
  const int tid = threadIdx.x;
  const float* s = S + (size_t)row * n;

  float mx = -3.4e38f;
  for (int c = tid; c < n; c += 256) mx = fmaxf(mx, s[c]);
  red[tid] = mx;
  __syncthreads();
  for (int st = 128; st > 0; st >>= 1) {
    if (tid < st) red[tid] = fmaxf(red[tid], red[tid + st]);
    __syncthreads();
  }
  mx = red[0];
  __syncthreads();

  float sum = 0.0f;
  for (int c = tid; c < n; c += 256) sum += expf(s[c] - mx);
  red[tid] = sum;
  __syncthreads();
  for (int st = 128; st > 0; st >>= 1) {
    if (tid < st) red[tid] += red[tid + st];
    __syncthreads();
  }
  const float inv = 1.0f / red[0];

  _Float16* p = P + (size_t)row * n;
  for (int c = tid; c < n; c += 256)
    p[c] = (_Float16)(expf(s[c] - mx) * inv);
}

// ---------------------------------------------------------------------------
// Tiled WMMA GEMM: C[M,N] = act(scale * (A[M,K] x B) + bias)
//   A: f16 row-major, lda = K  (staged via Tensor Data Mover)
//   BMODE 0: B f32 row-major [K,N] (weights; f32->f16 convert in staging)
//   BMODE 1: B f16 row-major [K,N] (activations, e.g. V; transposed staging)
//   BMODE 2: B f16 n-major  [N,K] (e.g. K-matrix for Q*K^T; staged via TDM)
//   NCHECK: guard N (vocab projection, N not a multiple of 128)
// 256 threads = 8 waves; block tile 128x128; wave computes 16 rows x 128 cols
// via 8x v_wmma_f32_16x16x32_f16 per K-step of 32.
// ---------------------------------------------------------------------------
template <int BMODE, bool GELU, bool NCHECK>
__global__ __launch_bounds__(256) void gemm_wmma_f16(
    const _Float16* __restrict__ A, const void* __restrict__ Bv,
    const float* __restrict__ bias, float* __restrict__ outF,
    _Float16* __restrict__ outH, int M, int N, int K, float scale) {
  __shared__ __align__(32) _Float16 As[TM][TK];  // row-major tile of A
  __shared__ __align__(32) _Float16 Bs[TN][TK];  // n-major tile of B (B^T)

  const int tid  = threadIdx.x;
  const int lane = tid & 31;
  const int wv   = tid >> 5;       // 0..7
  const int lrow = lane & 15;      // 0..15
  const int lhi  = lane >> 4;      // 0 or 1
  const int n0   = blockIdx.x * TN;
  const int m0   = blockIdx.y * TM;

  const unsigned ldsA = (unsigned)(uintptr_t)&As[0][0];
  const unsigned ldsB = (unsigned)(uintptr_t)&Bs[0][0];

  v8f acc[8] = {};

  for (int k0 = 0; k0 < K; k0 += TK) {
    // ---- stage A tile via TDM: 128 rows x 32 halves, row stride K ----
    if (wv == 0) {
      tdm_load_tile_f16(ldsA, A + (size_t)m0 * K + k0,
                        (unsigned)(K - k0), (unsigned)(M - m0),
                        TK, TM, (unsigned)K);
      if constexpr (BMODE == 2) {
        // B is n-major [N,K]: same 2D-tile shape as A
        tdm_load_tile_f16(ldsB, (const _Float16*)Bv + (size_t)n0 * K + k0,
                          (unsigned)(K - k0), (unsigned)(N - n0),
                          TK, TN, (unsigned)K);
      }
    }
    // ---- stage B tile into Bs[n][k] (n-major) where TDM can't ----
    if constexpr (BMODE == 1) {
      const _Float16* B = (const _Float16*)Bv;
      const int k  = tid >> 3;         // 0..31
      const int nc = (tid & 7) * 16;   // 0..112
      const _Float16* src = B + (size_t)(k0 + k) * N + n0 + nc;
#pragma unroll
      for (int i = 0; i < 16; ++i) Bs[nc + i][k] = src[i];
      if (k0 + TK < K)
        __builtin_prefetch(src + (size_t)TK * N, 0, 0);  // global_prefetch_b8
    } else if constexpr (BMODE == 0) {
      const float* B = (const float*)Bv;
      const int k  = tid >> 3;
      const int nc = (tid & 7) * 16;
      const float* src = B + (size_t)(k0 + k) * N + n0 + nc;
#pragma unroll
      for (int i = 0; i < 16; ++i) {
        _Float16 h = (_Float16)0.0f;
        if (!NCHECK || (n0 + nc + i) < N) h = (_Float16)src[i];
        Bs[nc + i][k] = h;
      }
      if (k0 + TK < K)
        __builtin_prefetch(src + (size_t)TK * N, 0, 0);
    }
    if (wv == 0) __builtin_amdgcn_s_wait_tensorcnt(0);
    __syncthreads();

    // ---- A fragment (16x32 f16 layout): lanes 0-15 K0-15,
    //      lanes 16-31 K16-31 for row M = lane%16 ----
    const v16h a = *(const v16h*)&As[wv * 16 + lrow][lhi * 16];
#pragma unroll
    for (int t = 0; t < 8; ++t) {
      // B fragment (32x16): lane = column, lanes 0-15 K0-15, 16-31 K16-31
      const v16h b = *(const v16h*)&Bs[t * 16 + lrow][lhi * 16];
      acc[t] = __builtin_amdgcn_wmma_f32_16x16x32_f16(
          false, a, false, b, (short)0, acc[t], false, false);
    }
    __syncthreads();
  }

  // ---- epilogue: scale, bias, GELU, store f32/f16 ----
#pragma unroll
  for (int t = 0; t < 8; ++t) {
    const int col = n0 + t * 16 + lrow;
    const bool cok = !NCHECK || (col < N);
    const float bval = (bias != nullptr && cok) ? bias[col] : 0.0f;
#pragma unroll
    for (int i = 0; i < 8; ++i) {
      const int row = m0 + wv * 16 + lhi * 8 + i;
      float v = acc[t][i] * scale + bval;
      if (GELU) v = 0.5f * v * (1.0f + erff(v * 0.70710678118f));
      if (cok) {
        if (outF) outF[(size_t)row * N + col] = v;
        if (outH) outH[(size_t)row * N + col] = (_Float16)v;
      }
    }
  }
}

// ---------------------------------------------------------------------------
// Host-side orchestration
// ---------------------------------------------------------------------------
extern "C" void kernel_launch(void* const* d_in, const int* in_sizes, int n_in,
                              void* d_out, int out_size, void* d_ws,
                              size_t ws_size, hipStream_t stream) {
  (void)in_sizes; (void)n_in; (void)out_size; (void)ws_size;

  const int*   ids  = (const int*)d_in[0];
  const float* emb  = (const float*)d_in[1];
  const float* Wq   = (const float*)d_in[2];
  const float* bq   = (const float*)d_in[3];
  const float* Wk   = (const float*)d_in[4];
  const float* bk   = (const float*)d_in[5];
  const float* Wv   = (const float*)d_in[6];
  const float* bv   = (const float*)d_in[7];
  const float* W1   = (const float*)d_in[8];
  const float* b1   = (const float*)d_in[9];
  const float* W2   = (const float*)d_in[10];
  const float* b2   = (const float*)d_in[11];
  const float* Wout = (const float*)d_in[12];
  const float* bout = (const float*)d_in[13];
  float* logits = (float*)d_out;

  // workspace layout (bytes)
  char* ws = (char*)d_ws;
  _Float16* xf16 = (_Float16*)(ws + 0);                    //  4 MB
  _Float16* qf16 = (_Float16*)(ws + (4u << 20));           //  4 MB
  _Float16* kf16 = (_Float16*)(ws + (8u << 20));           //  4 MB
  _Float16* vf16 = (_Float16*)(ws + (12u << 20));          //  4 MB
  _Float16* hf16 = (_Float16*)(ws + (16u << 20));          // 16 MB
  _Float16* prob = (_Float16*)(ws + (32u << 20));          //  8 MB
  float*    scor = (float*)(ws + (40u << 20));             // 16 MB

  const dim3 blk(256);
  const dim3 gD(D_MODEL / TN, SEQ / TM);   // (8, 16)
  const dim3 gS(SEQ / TN, SEQ / TM);       // (16, 16)
  const dim3 gF(D_FF / TN, SEQ / TM);      // (32, 16)
  const dim3 gV((VOCAB + TN - 1) / TN, SEQ / TM);  // (393, 16)
  const float att_scale = 0.03125f;  // 1/sqrt(1024)

  // embed
  embed_gather_f16<<<SEQ, blk, 0, stream>>>(ids, emb, xf16);

  for (int l = 0; l < N_LAYERS; ++l) {
    const float* Wql = Wq + (size_t)l * D_MODEL * D_MODEL;
    const float* Wkl = Wk + (size_t)l * D_MODEL * D_MODEL;
    const float* Wvl = Wv + (size_t)l * D_MODEL * D_MODEL;
    const float* W1l = W1 + (size_t)l * D_MODEL * D_FF;
    const float* W2l = W2 + (size_t)l * D_FF * D_MODEL;

    // q,k,v = x @ W + b  (f32 weights converted in staging)
    gemm_wmma_f16<0, false, false><<<gD, blk, 0, stream>>>(
        xf16, Wql, bq + (size_t)l * D_MODEL, nullptr, qf16,
        SEQ, D_MODEL, D_MODEL, 1.0f);
    gemm_wmma_f16<0, false, false><<<gD, blk, 0, stream>>>(
        xf16, Wkl, bk + (size_t)l * D_MODEL, nullptr, kf16,
        SEQ, D_MODEL, D_MODEL, 1.0f);
    gemm_wmma_f16<0, false, false><<<gD, blk, 0, stream>>>(
        xf16, Wvl, bv + (size_t)l * D_MODEL, nullptr, vf16,
        SEQ, D_MODEL, D_MODEL, 1.0f);

    // scores = (q @ k^T) * scale   (k is n-major wrt scores' N dim; TDM path)
    gemm_wmma_f16<2, false, false><<<gS, blk, 0, stream>>>(
        qf16, kf16, nullptr, scor, nullptr, SEQ, SEQ, D_MODEL, att_scale);

    // probs = softmax(scores)
    softmax_rows_f16<<<SEQ, blk, 0, stream>>>(scor, prob, SEQ);

    // x = probs @ v
    gemm_wmma_f16<1, false, false><<<gD, blk, 0, stream>>>(
        prob, vf16, nullptr, nullptr, xf16, SEQ, D_MODEL, SEQ, 1.0f);

    // h = gelu(x @ W1 + b1)
    gemm_wmma_f16<0, true, false><<<gF, blk, 0, stream>>>(
        xf16, W1l, b1 + (size_t)l * D_FF, nullptr, hf16,
        SEQ, D_FF, D_MODEL, 1.0f);

    // x = h @ W2 + b2
    gemm_wmma_f16<0, false, false><<<gD, blk, 0, stream>>>(
        hf16, W2l, b2 + (size_t)l * D_MODEL, nullptr, xf16,
        SEQ, D_MODEL, D_FF, 1.0f);
  }

  // logits = x @ Wout + bout  (N = 50257, bounds-checked)
  gemm_wmma_f16<0, false, true><<<gV, blk, 0, stream>>>(
      xf16, Wout, bout, logits, nullptr, SEQ, VOCAB, D_MODEL, 1.0f);
}
